// EncoderLayer_24764781429446
// MI455X (gfx1250) — compile-verified, matrix operand order checked by reading, not run on hardware
//
#include <hip/hip_runtime.h>

typedef unsigned short u16;
typedef __attribute__((ext_vector_type(16))) __bf16 v16bf;
typedef __attribute__((ext_vector_type(8)))  float  v8f;

#define NNODES 8192
#define KNBR   48
#define CDIM   128
#define HDIM   512

// ---------- helpers ----------
__device__ __forceinline__ u16 bh(float x) {             // f32 -> bf16 (RNE)
  union { float f; unsigned u; } v; v.f = x;
  unsigned r = v.u + 0x7FFFu + ((v.u >> 16) & 1u);
  return (u16)(r >> 16);
}
__device__ __forceinline__ float gelu_f(float x) {       // exact gelu
  return 0.5f * x * (1.0f + erff(x * 0.70710678118654752f));
}

// 16x32 bf16 fragment load, row-major source [*, stride] (halfs).
// A layout: lanes 0-15 row m=lane, K = [0..7,16..23]; lanes 16-31 same rows,
// K offset +8 ([8..15,24..31]).  B (as W rows = columns of B) uses the
// mirrored layout with n = lane&15.
__device__ __forceinline__ v16bf frag16x32(const u16* base, int stride, int k0, int lane) {
  const u16* p = base + (size_t)(lane & 15) * stride + k0 + ((lane >> 4) << 3);
  union { v16bf v; uint4 q[2]; } u;
  u.q[0] = *(const uint4*)(p);
  u.q[1] = *(const uint4*)(p + 16);
  return u.v;
}

__device__ __forceinline__ v8f wmma_bf16(v16bf a, v16bf b, v8f c) {
  return __builtin_amdgcn_wmma_f32_16x16x32_bf16(false, a, false, b, (short)0, c, false, false);
}

// CDNA5 async global->LDS copy (per-lane 16B), tracked by ASYNCcnt.
__device__ __forceinline__ void async_b128_to_lds(u16* lds, const u16* gsrc) {
  unsigned loff = (unsigned)(size_t)(void*)lds;          // low 32 bits = LDS byte addr
  asm volatile("global_load_async_to_lds_b128 %0, %1, off"
               :: "v"(loff), "v"(gsrc) : "memory");
}
__device__ __forceinline__ void wait_async0() {
  asm volatile("s_wait_asynccnt 0" ::: "memory");
}

// ---------- tiny conversion kernels ----------
__global__ void k_cvt(u16* __restrict__ dst, const float* __restrict__ src, int n) {
  for (int i = blockIdx.x * blockDim.x + threadIdx.x; i < n; i += gridDim.x * blockDim.x)
    dst[i] = bh(src[i]);
}

// w: [128, 384] -> three bf16 [128,128] blocks (cols 0:128 / 128:256 / 256:384)
__global__ void k_split3(u16* __restrict__ a, u16* __restrict__ b, u16* __restrict__ c,
                         const float* __restrict__ w) {
  int i = blockIdx.x * blockDim.x + threadIdx.x;
  if (i < 128 * 128) {
    int r = i >> 7, col = i & 127;
    const float* row = w + r * 384;
    a[i] = bh(row[col]); b[i] = bh(row[128 + col]); c[i] = bh(row[256 + col]);
  }
}

// ---------- generic WMMA GEMM: out = act(X @ W^T + bias [+ resid]) ----------
// X: [M,Kd] bf16, W: [Nd,Kd] bf16, block = 8 waves, wave -> one 16-col tile.
template<int ACT>
__global__ __launch_bounds__(256) void k_gemm(
    const u16* __restrict__ X, const u16* __restrict__ W,
    const float* __restrict__ bias, const float* __restrict__ resid,
    float* __restrict__ outF, u16* __restrict__ outB, int Kd, int Nd) {
  int lane = threadIdx.x & 31, wv = threadIdx.x >> 5;
  size_t m0 = (size_t)blockIdx.x * 16;
  int n0 = blockIdx.y * 128 + wv * 16;
  v8f acc = {0, 0, 0, 0, 0, 0, 0, 0};
  const u16* xb = X + m0 * Kd;
  const u16* wb = W + (size_t)n0 * Kd;
  for (int k0 = 0; k0 < Kd; k0 += 32) {
    v16bf a = frag16x32(xb, Kd, k0, lane);
    v16bf b = frag16x32(wb, Kd, k0, lane);
    acc = wmma_bf16(a, b, acc);
  }
  int n = n0 + (lane & 15), hi = lane >> 4;
#pragma unroll
  for (int r = 0; r < 8; ++r) {
    size_t m = m0 + r + hi * 8;
    float v = acc[r];
    if (bias)  v += bias[n];
    if (resid) v += resid[m * Nd + n];
    if (ACT)   v = gelu_f(v);
    if (outF)  outF[m * Nd + n] = v;
    if (outB)  outB[m * Nd + n] = bh(v);
  }
}

// ---------- LayerNorm over C=128 (1 row per block of 128 threads) ----------
// MODE 0: x = x + add*(1/30)   (post-message residual)
// MODE 1: y *= mask[row] after LN
template<int MODE>
__global__ __launch_bounds__(128) void k_ln(
    const float* __restrict__ x, const float* __restrict__ add,
    const float* __restrict__ mask, const float* __restrict__ g,
    const float* __restrict__ b, float* __restrict__ outF, u16* __restrict__ outB) {
  __shared__ float red[128];
  int row = blockIdx.x, n = threadIdx.x;
  size_t o = (size_t)row * 128 + n;
  float v = x[o];
  if (MODE == 0) v += add[o] * (1.0f / 30.0f);
  red[n] = v; __syncthreads();
  for (int s = 64; s; s >>= 1) { if (n < s) red[n] += red[n + s]; __syncthreads(); }
  float mu = red[0] * (1.0f / 128.0f); __syncthreads();
  float d = v - mu;
  red[n] = d * d; __syncthreads();
  for (int s = 64; s; s >>= 1) { if (n < s) red[n] += red[n + s]; __syncthreads(); }
  float var = red[0] * (1.0f / 128.0f);
  float y = d * rsqrtf(var + 1e-5f) * g[n] + b[n];
  if (MODE == 1) y *= mask[row];
  outF[o] = y;
  if (outB) outB[o] = bh(y);
}

// ---------- edge MLP kernel (msg / upd) ----------
// Block = 2 nodes x 48 edges = 96 rows.  3-layer 128->128 MLP, WMMA bf16,
// layer-0 row bias = hA[node] (incl. b0) + hC[neighbor] (L2 gather).
// MSG: write per-node sum over 48 edges.  UPD: residual(e) + LN3 -> e_out.
template<bool IS_UPD>
__global__ __launch_bounds__(256) void k_edge(
    const float* __restrict__ e, const int* __restrict__ nbr,
    const float* __restrict__ hA, const float* __restrict__ hC,
    const u16* __restrict__ W0b, const u16* __restrict__ W1,
    const u16* __restrict__ W2, const float* __restrict__ b1,
    const float* __restrict__ b2, const float* __restrict__ lng,
    const float* __restrict__ lnb, float* __restrict__ out) {
  __shared__ __align__(16) u16 sX[2][96 * CDIM];   // activation ping-pong (bf16)
  __shared__ __align__(16) u16 sW[CDIM * CDIM];    // current layer weights (bf16)
  __shared__ float sO[96 * 132];                   // final f32 (padded stride)
  __shared__ float sBias[CDIM];
  __shared__ float sA[2 * CDIM];
  __shared__ int   sN[96];
  __shared__ float sMu[96], sRs[96], sG[CDIM], sB[CDIM];

  int tid = threadIdx.x, lane = tid & 31, wv = tid >> 5;
  int node0 = blockIdx.x * 2;
  size_t ebase = (size_t)node0 * KNBR * CDIM;

  if (tid < 96) sN[tid] = nbr[node0 * KNBR + tid];
  sA[tid] = hA[(size_t)node0 * CDIM + tid];        // 2 rows x 128 = 256 = blockDim
  if (IS_UPD && tid < CDIM) { sG[tid] = lng[tid]; sB[tid] = lnb[tid]; }

  // stage e -> bf16 into X0
  for (int i = tid; i < 96 * CDIM / 4; i += 256) {
    float4 v = ((const float4*)(e + ebase))[i];
    ushort4 o; o.x = bh(v.x); o.y = bh(v.y); o.z = bh(v.z); o.w = bh(v.w);
    *(ushort4*)(&sX[0][i * 4]) = o;
  }

  const u16* Wl[3] = {W0b, W1, W2};
  int cur = 0;
  for (int l = 0; l < 3; ++l) {
    __syncthreads();
    // async-stage this layer's 128x128 bf16 weights into LDS
#pragma unroll
    for (int c = 0; c < 8; ++c) {
      int ch = c * 256 + tid;                      // 2048 chunks x 16B = 32KB
      async_b128_to_lds(&sW[ch * 8], Wl[l] + ch * 8);
    }
    if (l == 1 && tid < CDIM) sBias[tid] = b1[tid];
    if (l == 2 && tid < CDIM) sBias[tid] = b2[tid];
    wait_async0();
    __syncthreads();

    int n0 = wv * 16;
    v16bf bfr[4];
#pragma unroll
    for (int ks = 0; ks < 4; ++ks) bfr[ks] = frag16x32(sW + n0 * CDIM, CDIM, ks * 32, lane);

#pragma unroll
    for (int rt = 0; rt < 6; ++rt) {
      v8f acc = {0, 0, 0, 0, 0, 0, 0, 0};
#pragma unroll
      for (int ks = 0; ks < 4; ++ks) {
        v16bf a = frag16x32(&sX[cur][rt * 16 * CDIM], CDIM, ks * 32, lane);
        acc = wmma_bf16(a, bfr[ks], acc);
      }
      int n = n0 + (lane & 15), hi = lane >> 4;
#pragma unroll
      for (int r = 0; r < 8; ++r) {
        int m = rt * 16 + r + hi * 8;
        float v = acc[r];
        if (l == 0) v += sA[((m >= KNBR) ? CDIM : 0) + n] + hC[(size_t)sN[m] * CDIM + n];
        else        v += sBias[n];
        if (l < 2) { v = gelu_f(v); sX[cur ^ 1][m * CDIM + n] = bh(v); }
        else       { sO[m * 132 + n] = v; }
      }
    }
    cur ^= 1;
  }
  __syncthreads();

  if (!IS_UPD) {
    // sum messages over 48 edges: 2 nodes x 128 cols = 256 outputs
    int nd = tid >> 7, n = tid & 127;
    float s = 0.f;
#pragma unroll 8
    for (int r = 0; r < KNBR; ++r) s += sO[(nd * KNBR + r) * 132 + n];
    out[(size_t)(node0 + nd) * CDIM + n] = s;
  } else {
    // e residual + LayerNorm per edge row
    for (int i = tid; i < 96 * CDIM; i += 256) {
      int m = i >> 7, n = i & 127;
      sO[m * 132 + n] += e[ebase + i];
    }
    __syncthreads();
    if (tid < 96) {
      float s = 0.f, s2 = 0.f;
      for (int n = 0; n < CDIM; ++n) { float v = sO[tid * 132 + n]; s += v; s2 += v * v; }
      float mu = s * (1.0f / 128.0f);
      sMu[tid] = mu;
      sRs[tid] = rsqrtf(s2 * (1.0f / 128.0f) - mu * mu + 1e-5f);
    }
    __syncthreads();
    for (int i = tid; i < 96 * CDIM; i += 256) {
      int m = i >> 7, n = i & 127;
      out[ebase + i] = (sO[m * 132 + n] - sMu[m]) * sRs[m] * sG[n] + sB[n];
    }
  }
}

// ---------- host ----------
extern "C" void kernel_launch(void* const* d_in, const int* in_sizes, int n_in,
                              void* d_out, int out_size, void* d_ws, size_t ws_size,
                              hipStream_t stream) {
  (void)in_sizes; (void)n_in; (void)out_size; (void)ws_size;
  const float* h    = (const float*)d_in[0];
  const float* e    = (const float*)d_in[1];
  const int*   nbr  = (const int*)  d_in[2];
  const float* mask = (const float*)d_in[3];
  const float* mW0  = (const float*)d_in[4];
  const float* mb0  = (const float*)d_in[5];
  const float* mW1  = (const float*)d_in[6];
  const float* mb1  = (const float*)d_in[7];
  const float* mW2  = (const float*)d_in[8];
  const float* mb2  = (const float*)d_in[9];
  const float* l1g  = (const float*)d_in[10];
  const float* l1b  = (const float*)d_in[11];
  const float* dW0  = (const float*)d_in[12];
  const float* db0  = (const float*)d_in[13];
  const float* dW1  = (const float*)d_in[14];
  const float* db1  = (const float*)d_in[15];
  const float* l2g  = (const float*)d_in[16];
  const float* l2b  = (const float*)d_in[17];
  const float* uW0  = (const float*)d_in[18];
  const float* ub0  = (const float*)d_in[19];
  const float* uW1  = (const float*)d_in[20];
  const float* ub1  = (const float*)d_in[21];
  const float* uW2  = (const float*)d_in[22];
  const float* ub2  = (const float*)d_in[23];
  const float* l3g  = (const float*)d_in[24];
  const float* l3b  = (const float*)d_in[25];

  char* w = (char*)d_ws;
  auto take = [&](size_t bytes) -> char* {
    char* p = w; w += (bytes + 255) & ~(size_t)255; return p;
  };
  u16*   h_bf = (u16*)  take((size_t)NNODES * CDIM * 2);
  float* hAm  = (float*)take((size_t)NNODES * CDIM * 4);
  float* hCm  = (float*)take((size_t)NNODES * CDIM * 4);
  float* msum = (float*)take((size_t)NNODES * CDIM * 4);
  float* h1f  = (float*)take((size_t)NNODES * CDIM * 4);
  u16*   h1b  = (u16*)  take((size_t)NNODES * CDIM * 2);
  u16*   d0b  = (u16*)  take((size_t)NNODES * HDIM * 2);
  float* h2f  = (float*)take((size_t)NNODES * CDIM * 4);
  u16*   h3b  = (u16*)  take((size_t)NNODES * CDIM * 2);
  float* hAu  = (float*)take((size_t)NNODES * CDIM * 4);
  float* hCu  = (float*)take((size_t)NNODES * CDIM * 4);
  u16* mA   = (u16*)take(128 * 128 * 2);
  u16* mB   = (u16*)take(128 * 128 * 2);
  u16* mC   = (u16*)take(128 * 128 * 2);
  u16* mW1b = (u16*)take(128 * 128 * 2);
  u16* mW2b = (u16*)take(128 * 128 * 2);
  u16* dW0b = (u16*)take(HDIM * CDIM * 2);
  u16* dW1b = (u16*)take(CDIM * HDIM * 2);
  u16* uA   = (u16*)take(128 * 128 * 2);
  u16* uB   = (u16*)take(128 * 128 * 2);
  u16* uC   = (u16*)take(128 * 128 * 2);
  u16* uW1b = (u16*)take(128 * 128 * 2);
  u16* uW2b = (u16*)take(128 * 128 * 2);

  // conversions
  k_cvt<<<1024, 256, 0, stream>>>(h_bf, h, NNODES * CDIM);
  k_split3<<<64, 256, 0, stream>>>(mA, mB, mC, mW0);
  k_split3<<<64, 256, 0, stream>>>(uA, uB, uC, uW0);
  k_cvt<<<64, 256, 0, stream>>>(mW1b, mW1, 128 * 128);
  k_cvt<<<64, 256, 0, stream>>>(mW2b, mW2, 128 * 128);
  k_cvt<<<256, 256, 0, stream>>>(dW0b, dW0, HDIM * CDIM);
  k_cvt<<<256, 256, 0, stream>>>(dW1b, dW1, CDIM * HDIM);
  k_cvt<<<64, 256, 0, stream>>>(uW1b, uW1, 128 * 128);
  k_cvt<<<64, 256, 0, stream>>>(uW2b, uW2, 128 * 128);

  dim3 g1(NNODES / 16, 1);
  // per-node precompute for msg MLP: hA = h@W0a^T + b0, hC = h@W0c^T
  k_gemm<0><<<g1, 256, 0, stream>>>(h_bf, mA, mb0, nullptr, hAm, nullptr, CDIM, CDIM);
  k_gemm<0><<<g1, 256, 0, stream>>>(h_bf, mC, nullptr, nullptr, hCm, nullptr, CDIM, CDIM);

  // edge message MLP + sum over K
  k_edge<false><<<NNODES / 2, 256, 0, stream>>>(e, nbr, hAm, hCm, mB, mW1b, mW2b,
                                                mb1, mb2, nullptr, nullptr, msum);

  // h1 = LN1(h + msum/30)
  k_ln<0><<<NNODES, 128, 0, stream>>>(h, msum, nullptr, l1g, l1b, h1f, h1b);

  // dense block: h2 = h1 + gelu(h1@dW0^T + db0)@dW1^T + db1
  k_gemm<1><<<dim3(NNODES / 16, HDIM / 128), 256, 0, stream>>>(
      h1b, dW0b, db0, nullptr, nullptr, d0b, CDIM, HDIM);
  k_gemm<0><<<g1, 256, 0, stream>>>(d0b, dW1b, db1, h1f, h2f, nullptr, HDIM, CDIM);

  float* hout = (float*)d_out;
  float* eout = hout + (size_t)NNODES * CDIM;
  // h = mask * LN2(h2)  -> output h + bf16 copy
  k_ln<1><<<NNODES, 128, 0, stream>>>(h2f, nullptr, mask, l2g, l2b, hout, h3b);

  // per-node precompute for upd MLP (on updated h)
  k_gemm<0><<<g1, 256, 0, stream>>>(h3b, uA, ub0, nullptr, hAu, nullptr, CDIM, CDIM);
  k_gemm<0><<<g1, 256, 0, stream>>>(h3b, uC, nullptr, nullptr, hCu, nullptr, CDIM, CDIM);

  // edge update MLP + residual + LN3 -> e output
  k_edge<true><<<NNODES / 2, 256, 0, stream>>>(e, nbr, hAu, hCu, uB, uW1b, uW2b,
                                               ub1, ub2, l3g, l3b, eout);
}
